// SelfEfficientGlobalPointerV2_10977936409144
// MI455X (gfx1250) — compile-verified, matrix operand order checked by reading, not run on hardware
//
#include <hip/hip_runtime.h>

// CDNA5 / gfx1250, wave32. WMMA f32 = A(f16 16x32) x B(f16 32x16) + C(f32 16x16).
typedef __attribute__((ext_vector_type(16))) _Float16 v16h;
typedef __attribute__((ext_vector_type(8)))  _Float16 v8h;
typedef __attribute__((ext_vector_type(8)))  float    v8f;

#define NB      8
#define NSEQ    1024
#define HID     1024
#define NHEADS  12

union ABfrag { v16h v; v8h h[2]; };

// ---------------------------------------------------------------------------
// inputs f32 -> f16 (one shot; 32MB reads, 16MB writes). Removes the need for
// in-GEMM conversion + LDS staging, so the GEMM loop is pure b128 loads+WMMA.
// ---------------------------------------------------------------------------
__global__ void __launch_bounds__(256)
cvt_inputs_kernel(const float* __restrict__ in, _Float16* __restrict__ out) {
  size_t i = ((size_t)blockIdx.x * 256 + threadIdx.x) * 8;
  const float4* p = (const float4*)(in + i);
  float4 a = p[0];
  float4 b = p[1];
  v8h o;
  o[0] = (_Float16)a.x; o[1] = (_Float16)a.y;
  o[2] = (_Float16)a.z; o[3] = (_Float16)a.w;
  o[4] = (_Float16)b.x; o[5] = (_Float16)b.y;
  o[6] = (_Float16)b.z; o[7] = (_Float16)b.w;
  *(v8h*)(out + i) = o;
}

// ---------------------------------------------------------------------------
// Pack W1 (1024x128) and W2 (1024x24, padded to 32) into WMMA B-fragment
// order: [kchunk 0..31][coltile 0..9][lane 0..31][half 0..15], f16.
// Half->K mapping (16-bit operands, wave32):
//   lanes 0-15 : K = (h<8 ? h : h+8)        in {0..7, 16..23}
//   lanes 16-31: K = (h<8 ? h : h+8) + 8    in {8..15, 24..31}
// ---------------------------------------------------------------------------
__global__ void pack_w_kernel(const float* __restrict__ W1,
                              const float* __restrict__ W2,
                              _Float16* __restrict__ wpack) {
  int lane = threadIdx.x;            // 32
  int kc   = blockIdx.x;             // 32 k-chunks of 32
  int tile = blockIdx.y;             // 0..7 -> W1, 8..9 -> W2
  int hi   = (lane >= 16) ? 8 : 0;
  int coll = lane & 15;
  _Float16* dst = wpack + (((size_t)kc * 10 + tile) * 32 + lane) * 16;
#pragma unroll
  for (int h = 0; h < 16; ++h) {
    int kl = ((h < 8) ? h : h + 8) + hi;
    int k  = kc * 32 + kl;
    float v;
    if (tile < 8) {
      v = W1[(size_t)k * 128 + tile * 16 + coll];
    } else {
      int c = (tile - 8) * 16 + coll;
      v = (c < 24) ? W2[(size_t)k * 24 + c] : 0.0f;
    }
    dst[h] = (_Float16)v;
  }
}

// ---------------------------------------------------------------------------
// Projection + RoPE + bias kernel.
// Block = 320 threads (10 waves), one 16-row tile of the flattened (B*N) rows.
// Waves 0-7: x = inputs@W1 cols [16w,16w+16) ; waves 8-9: inputs@W2 (24 cols).
// A fragments loaded straight from the pre-converted f16 inputs (two b128s
// per lane per k-step, no LDS, no barriers in the K loop); B fragments from
// the packed weights. One WMMA per wave per 32-wide k-step.
// Epilogue: +b1, RoPE split even/odd -> qw/kw f16; +b2, /2 -> biasw f32.
// ---------------------------------------------------------------------------
__global__ void __launch_bounds__(320)
proj_rope_kernel(const _Float16* __restrict__ inf16,
                 const float* __restrict__ b1,
                 const float* __restrict__ b2,
                 const _Float16* __restrict__ wpack,
                 _Float16* __restrict__ qw,
                 _Float16* __restrict__ kw,
                 float* __restrict__ biasw) {
  __shared__ float xbuf[16 * 132];         // x tile (+b1), padded stride

  const int tid  = threadIdx.x;
  const int wave = tid >> 5;
  const int lane = tid & 31;
  const int row0 = blockIdx.x * 16;        // flattened row (b*1024 + n)

  const int arow  = lane & 15;             // A: M = lane%16
  const int abase = (lane >> 4) * 8;       // K-half offset for upper lanes

  const _Float16* ap = inf16 + (size_t)(row0 + arow) * HID + abase;
  const _Float16* wp = wpack + ((size_t)wave * 32 + lane) * 16;

  v8f c = {};
#pragma unroll 4
  for (int kc = 0; kc < 32; ++kc) {
    ABfrag a, bf;
    a.h[0] = *(const v8h*)(ap + kc * 32);        // K base .. base+7
    a.h[1] = *(const v8h*)(ap + kc * 32 + 16);   // K base+16 .. base+23
    const _Float16* w = wp + (size_t)kc * (10 * 32 * 16);
    bf.h[0] = *(const v8h*)(w);
    bf.h[1] = *(const v8h*)(w + 8);
    c = __builtin_amdgcn_wmma_f32_16x16x32_f16(false, a.v, false, bf.v,
                                               (short)0, c, false, false);
  }

  const int rbase = (lane >> 4) * 8;       // C: M = r + 8*(lane>=16)
  if (wave < 8) {
    int col = wave * 16 + (lane & 15);
    float bb = b1[col];
#pragma unroll
    for (int r = 0; r < 8; ++r)
      xbuf[(r + rbase) * 132 + col] = c[r] + bb;
  } else {
    int col = (wave - 8) * 16 + (lane & 15);
    if (col < 24) {
      float bb = b2[col];
#pragma unroll
      for (int r = 0; r < 8; ++r)
        biasw[(size_t)(row0 + r + rbase) * 24 + col] = (c[r] + bb) * 0.5f;
    }
  }
  __syncthreads();

  // RoPE: qw[j]=x[2j], kw[j]=x[2j+1]; partner at x col +/-2; angle pos*inv[j/2]
  for (int i = tid; i < 16 * 128; i += 320) {
    int r    = i >> 7;
    int ccol = i & 127;
    int grow = row0 + r;
    int pos  = grow & (NSEQ - 1);
    int j    = ccol >> 1;
    int p    = j >> 1;
    // inv = 10000^(-2p/64) = exp(-p * ln(10000)/32)
    float inv = __expf(-0.28782313662425572f * (float)p);
    float s, co;
    __sincosf((float)pos * inv, &s, &co);
    float v = xbuf[r * 132 + ccol];
    float partner = (j & 1) ? xbuf[r * 132 + ccol - 2]
                            : -xbuf[r * 132 + ccol + 2];
    _Float16 o = (_Float16)(v * co + partner * s);
    if (ccol & 1) kw[(size_t)grow * 64 + j] = o;
    else          qw[(size_t)grow * 64 + j] = o;
  }
}

// ---------------------------------------------------------------------------
// Logits kernel: qk = qw.kw^T / 8 via 2x WMMA (K=64), then broadcast across
// 12 heads with per-col even-bias, per-row odd-bias, mask, causal -1e12.
// Block = 256 threads (8 waves); block tile = 32 rows x 64 cols; wave = 16x16.
// Store-bandwidth bound: one qk tile amortized over 12 head planes.
// ---------------------------------------------------------------------------
__global__ void __launch_bounds__(256)
logits_kernel(const _Float16* __restrict__ qw,
              const _Float16* __restrict__ kw,
              const float* __restrict__ biasw,
              const int* __restrict__ mask,
              float* __restrict__ out) {
  __shared__ float bE[64 * 12];   // bias[b, 2h,   n0+j]  (added along cols)
  __shared__ float bO[32 * 12];   // bias[b, 2h+1, m0+i]  (added along rows)
  __shared__ float mR[32];
  __shared__ float mC[64];

  const int tid = threadIdx.x;
  const int n0  = blockIdx.x * 64;
  const int m0  = blockIdx.y * 32;
  const int b   = blockIdx.z;

  for (int i = tid; i < 64 * 12; i += 256)
    bE[i] = biasw[(size_t)(b * NSEQ + n0 + i / 12) * 24 + 2 * (i % 12)];
  for (int i = tid; i < 32 * 12; i += 256)
    bO[i] = biasw[(size_t)(b * NSEQ + m0 + i / 12) * 24 + 2 * (i % 12) + 1];
  if (tid < 64) mC[tid] = (float)mask[b * NSEQ + n0 + tid];
  if (tid < 32) mR[tid] = (float)mask[b * NSEQ + m0 + tid];
  __syncthreads();

  const int wave = tid >> 5, lane = tid & 31;
  const int wm = wave >> 2, wn = wave & 3;
  const int hi8  = (lane >> 4) * 8;
  const int rowA = m0 + wm * 16 + (lane & 15);   // A frag: M = lane%16
  const int colB = n0 + wn * 16 + (lane & 15);   // B frag: N = lane%16

  const _Float16* qrow = qw + ((size_t)b * NSEQ + rowA) * 64;
  const _Float16* krow = kw + ((size_t)b * NSEQ + colB) * 64;
  ABfrag a0, a1, b0, b1f;
  a0.h[0]  = *(const v8h*)(qrow + hi8);
  a0.h[1]  = *(const v8h*)(qrow + hi8 + 16);
  a1.h[0]  = *(const v8h*)(qrow + 32 + hi8);
  a1.h[1]  = *(const v8h*)(qrow + 48 + hi8);
  b0.h[0]  = *(const v8h*)(krow + hi8);
  b0.h[1]  = *(const v8h*)(krow + hi8 + 16);
  b1f.h[0] = *(const v8h*)(krow + 32 + hi8);
  b1f.h[1] = *(const v8h*)(krow + 48 + hi8);

  v8f c = {};
  c = __builtin_amdgcn_wmma_f32_16x16x32_f16(false, a0.v, false, b0.v,
                                             (short)0, c, false, false);
  c = __builtin_amdgcn_wmma_f32_16x16x32_f16(false, a1.v, false, b1f.v,
                                             (short)0, c, false, false);

  const int colL = wn * 16 + (lane & 15);
  const int gcol = n0 + colL;
#pragma unroll
  for (int r = 0; r < 8; ++r) {
    int rowL = wm * 16 + r + hi8;
    int grow = m0 + rowL;
    float m2  = mR[rowL] * mC[colL];
    float trl = (grow > gcol) ? 1.0e12f : 0.0f;
    float q   = c[r] * 0.125f;                  // 1/sqrt(64)
    float negpart = -1.0e12f * (1.0f - m2) - trl;
    size_t base = ((size_t)grow << 10) + (size_t)gcol;
#pragma unroll
    for (int h = 0; h < NHEADS; ++h) {
      float lg = q + bE[colL * 12 + h] + bO[rowL * 12 + h];
      lg = lg * m2 + negpart;
      out[(((size_t)(b * NHEADS + h)) << 20) + base] = lg;
    }
  }
}

// ---------------------------------------------------------------------------
// tril_mask output: (m>n)?1:0 broadcast over B*HEADS planes; b128 stores.
// ---------------------------------------------------------------------------
__global__ void tril_kernel(float* __restrict__ trout) {
  size_t idx  = (size_t)blockIdx.x * blockDim.x + threadIdx.x;  // one float4
  size_t pidx = idx & (size_t)(262144 - 1);                     // 1024*1024/4
  int m  = (int)(pidx >> 8);
  int c4 = (int)(pidx & 255) * 4;
  float4 v;
  v.x = (m > c4)     ? 1.0f : 0.0f;
  v.y = (m > c4 + 1) ? 1.0f : 0.0f;
  v.z = (m > c4 + 2) ? 1.0f : 0.0f;
  v.w = (m > c4 + 3) ? 1.0f : 0.0f;
  ((float4*)trout)[idx] = v;
}

// ---------------------------------------------------------------------------
extern "C" void kernel_launch(void* const* d_in, const int* in_sizes, int n_in,
                              void* d_out, int out_size, void* d_ws, size_t ws_size,
                              hipStream_t stream) {
  (void)in_sizes; (void)n_in; (void)out_size; (void)ws_size;
  const float* inputs = (const float*)d_in[0];
  const int*   mask   = (const int*)  d_in[1];
  const float* W1     = (const float*)d_in[2];
  const float* b1     = (const float*)d_in[3];
  const float* W2     = (const float*)d_in[4];
  const float* b2     = (const float*)d_in[5];

  float* logits = (float*)d_out;
  const size_t logit_elems = (size_t)NB * NHEADS * NSEQ * NSEQ;  // 100,663,296
  float* trout  = logits + logit_elems;

  char* ws = (char*)d_ws;
  _Float16* inf16 = (_Float16*)(ws);                  // 16 MB
  _Float16* qw    = (_Float16*)(ws + (16u << 20));    // 1 MB
  _Float16* kw    = (_Float16*)(ws + (17u << 20));    // 1 MB
  float*    biasw = (float*)   (ws + (18u << 20));    // 768 KB
  _Float16* wpack = (_Float16*)(ws + (19u << 20));    // 320 KB

  cvt_inputs_kernel<<<4096,            256, 0, stream>>>(inputs, inf16);
  pack_w_kernel    <<<dim3(32, 10),    32,  0, stream>>>(W1, W2, wpack);
  proj_rope_kernel <<<512,             320, 0, stream>>>(inf16, b1, b2, wpack,
                                                         qw, kw, biasw);
  logits_kernel    <<<dim3(16, 32, 8), 256, 0, stream>>>(qw, kw, biasw, mask, logits);
  tril_kernel      <<<98304,           256, 0, stream>>>(trout);
}
